// MultiHeadSelfAttention_12824772346578
// MI455X (gfx1250) — compile-verified, matrix operand order checked by reading, not run on hardware
//
#include <hip/hip_runtime.h>
#include <hip/hip_bf16.h>

// ---------------------------------------------------------------------------
// Multi-head causal self-attention, MI455X (gfx1250, wave32).
// All GEMMs via v_wmma_f32_16x16x32_bf16 (f32 accumulate).
// Flash-attention: scores never leave registers/LDS.
// V is stored transposed [b,h,hd,t] so every WMMA fragment load is a pair of
// contiguous 128-bit loads (no scalar gathers anywhere).
// ---------------------------------------------------------------------------

#define BATCH 2
#define SEQ   2048
#define DIMSZ 1024
#define NHEAD 16
#define HD    64
#define BT    (BATCH * SEQ)   // 4096 rows
#define ND    (3 * DIMSZ)     // 3072 qkv cols
#define NEG_INF (-__builtin_huge_valf())

typedef __bf16 bf16;
typedef __attribute__((ext_vector_type(16))) __bf16 v16bf;
typedef __attribute__((ext_vector_type(8)))  __bf16 v8bf;
typedef __attribute__((ext_vector_type(8)))  float  v8f;

// ---- WMMA helper -----------------------------------------------------------
__device__ __forceinline__ v8f wmma_bf16(v16bf a, v16bf b, v8f c) {
  // 8 args: (neg_a, A, neg_b, B, c_mod, C, reuse_a, reuse_b)
  return __builtin_amdgcn_wmma_f32_16x16x32_bf16(false, a, false, b,
                                                 (short)0, c, false, false);
}

// A/B fragment, 16x32 bf16, contiguous-K source (row stride ld elements).
// CDNA5 16-bit layout: lane m (0-15) = row/col m; half = lane>>4 selects
// K-chunks {half*8 .. half*8+7} and {16+half*8 .. 16+half*8+7}.
__device__ __forceinline__ v16bf load_frag(const bf16* p, int ld) {
  const int lane = threadIdx.x & 31;
  const bf16* base = p + (lane & 15) * ld + (lane >> 4) * 8;
  v8bf lo = *(const v8bf*)(base);
  v8bf hi = *(const v8bf*)(base + 16);
  return __builtin_shufflevector(lo, hi,
      0, 1, 2, 3, 4, 5, 6, 7, 8, 9, 10, 11, 12, 13, 14, 15);
}

// ---- f32 -> bf16 cast ------------------------------------------------------
__global__ void mhsa_cast_bf16(const float* __restrict__ s,
                               bf16* __restrict__ d, int n) {
  int i  = blockIdx.x * blockDim.x + threadIdx.x;
  int st = gridDim.x * blockDim.x;
  for (; i < n; i += st) d[i] = (bf16)s[i];
}

// ---- QKV projection: [BT,D] x [ND,D]^T, scatter to head layouts ------------
// One wave computes a 64x64 output tile. Q gets 1/sqrt(hd) folded in.
// Q,K -> [b,h,t,hd] ; V -> transposed [b,h,hd,t] for gather-free P@V frags.
// Scatter is fully select-based (no EXEC-masked branches).
__global__ __launch_bounds__(256) void mhsa_qkv_gemm(
    const bf16* __restrict__ X, const bf16* __restrict__ W,
    const float* __restrict__ bias,
    bf16* __restrict__ Q, bf16* __restrict__ K, bf16* __restrict__ Vt) {
  const int NT   = ND / 64;  // 48
  const int wave = (blockIdx.x * blockDim.x + threadIdx.x) >> 5;
  if (wave >= (BT / 64) * NT) return;
  const int m0 = (wave / NT) * 64, n0 = (wave % NT) * 64;
  const int lane = threadIdx.x & 31;
  const int coln = lane & 15, rbase = (lane >> 4) * 8;
  // 64-row tiles never straddle a batch boundary: batch index is wave-uniform.
  const int bb = m0 / SEQ;
  const int tb = m0 % SEQ;

  v8f acc[4][4] = {};
  for (int k = 0; k < DIMSZ; k += 32) {
    v16bf af[4], bfg[4];
#pragma unroll
    for (int i = 0; i < 4; ++i)
      af[i] = load_frag(X + (size_t)(m0 + i * 16) * DIMSZ + k, DIMSZ);
#pragma unroll
    for (int j = 0; j < 4; ++j)
      bfg[j] = load_frag(W + (size_t)(n0 + j * 16) * DIMSZ + k, DIMSZ);
#pragma unroll
    for (int i = 0; i < 4; ++i)
#pragma unroll
      for (int j = 0; j < 4; ++j)
        acc[i][j] = wmma_bf16(af[i], bfg[j], acc[i][j]);
  }

#pragma unroll
  for (int j = 0; j < 4; ++j) {
    const int n   = n0 + j * 16 + coln;
    const int sel = n >> 10;                 // /DIMSZ: 0=Q 1=K 2=V
    const int dc  = n & (DIMSZ - 1);
    const int h   = dc >> 6, hdc = dc & (HD - 1);
    const float bv  = bias[n];
    const float scl = (sel == 0) ? 0.125f : 1.0f;  // 1/sqrt(64)
    bf16* const dst = (sel == 0) ? Q : ((sel == 1) ? K : Vt);
    const size_t head = (size_t)(bb * NHEAD + h);
    const bool isv = (sel == 2);
#pragma unroll
    for (int i = 0; i < 4; ++i)
#pragma unroll
      for (int r = 0; r < 8; ++r) {
        const int t = tb + i * 16 + rbase + r;
        const size_t idx = isv ? (head * HD + hdc) * SEQ + t
                               : (head * SEQ + t) * HD + hdc;
        dst[idx] = (bf16)((acc[i][j][r] + bv) * scl);
      }
  }
}

// ---- Flash attention: one wave per (b, h, 64-row q block) ------------------
__global__ __launch_bounds__(256) void mhsa_flash_attn(
    const bf16* __restrict__ Q, const bf16* __restrict__ K,
    const bf16* __restrict__ Vt, const float* __restrict__ msk,
    bf16* __restrict__ O) {
  __shared__ bf16 Plds[8][64 * 64];          // 8 KB per wave, 64 KB total
  bf16* P = &Plds[threadIdx.x >> 5][0];

  const int wave = (blockIdx.x * blockDim.x + threadIdx.x) >> 5;
  const int qb = wave & (SEQ / 64 - 1);      // 0..31
  const int bh = wave >> 5;                  // 0..31
  const int bb = bh >> 4, h = bh & 15;
  const int q0 = qb * 64;
  const bf16* Qh = Q  + (size_t)(bb * NHEAD + h) * SEQ * HD;
  const bf16* Kh = K  + (size_t)(bb * NHEAD + h) * SEQ * HD;
  const bf16* Vh = Vt + (size_t)(bb * NHEAD + h) * HD * SEQ;  // [hd][t]
  const int lane = threadIdx.x & 31;
  const int coln = lane & 15, rbase = (lane >> 4) * 8;

  v16bf qf[4][2];
#pragma unroll
  for (int i = 0; i < 4; ++i)
#pragma unroll
    for (int kh = 0; kh < 2; ++kh)
      qf[i][kh] = load_frag(Qh + (size_t)(q0 + i * 16) * HD + kh * 32, HD);

  v8f o_acc[4][4] = {};
  float mst[4][8], lst[4][8];
#pragma unroll
  for (int i = 0; i < 4; ++i)
#pragma unroll
    for (int r = 0; r < 8; ++r) { mst[i][r] = NEG_INF; lst[i][r] = 0.0f; }

  const int nkb = qb + 1;                    // causal: only blocks <= diagonal
  for (int kb = 0; kb < nkb; ++kb) {
    const int k0 = kb * 64;

    // Prefetch next key block into WGP caches (global_prefetch_b8);
    // whole working set is L2-resident, this hides L2->WGP latency.
    if (kb + 1 < nkb) {
      __builtin_prefetch(Kh + (size_t)(k0 + 64) * HD + lane * 16, 0, 3);
      __builtin_prefetch(Vh + (size_t)lane * SEQ + k0 + 64, 0, 3);
    }

    v16bf kf[4][2];                          // K^T fragments (keys = columns)
#pragma unroll
    for (int ni = 0; ni < 4; ++ni)
#pragma unroll
      for (int kh = 0; kh < 2; ++kh)
        kf[ni][kh] = load_frag(Kh + (size_t)(k0 + ni * 16) * HD + kh * 32, HD);

    v16bf vf[2][4];                          // V frags [32 keys x 16 hd]
#pragma unroll
    for (int ki = 0; ki < 2; ++ki)
#pragma unroll
      for (int nj = 0; nj < 4; ++nj)
        vf[ki][nj] =
            load_frag(Vh + (size_t)(nj * 16) * SEQ + k0 + ki * 32, SEQ);

    float mv[4];
#pragma unroll
    for (int ni = 0; ni < 4; ++ni)
      mv[ni] = msk[bb * SEQ + k0 + ni * 16 + coln];

    for (int mi = 0; mi < 4; ++mi) {
      v8f s[4];
#pragma unroll
      for (int ni = 0; ni < 4; ++ni) {
        v8f a = {};
        a = wmma_bf16(qf[mi][0], kf[ni][0], a);
        a = wmma_bf16(qf[mi][1], kf[ni][1], a);
        s[ni] = a;
      }
      // causal + padding mask (select-based)
#pragma unroll
      for (int ni = 0; ni < 4; ++ni) {
        const int key = k0 + ni * 16 + coln;
#pragma unroll
        for (int r = 0; r < 8; ++r) {
          const int qrow = q0 + mi * 16 + rbase + r;
          s[ni][r] = (key > qrow || mv[ni] == 0.0f) ? NEG_INF : s[ni][r];
        }
      }
      // online softmax (row stats shared by the 16 lanes of each half-wave)
#pragma unroll
      for (int r = 0; r < 8; ++r) {
        float mx = fmaxf(fmaxf(s[0][r], s[1][r]), fmaxf(s[2][r], s[3][r]));
#pragma unroll
        for (int off = 1; off < 16; off <<= 1)
          mx = fmaxf(mx, __shfl_xor(mx, off, 32));
        const float mnew = fmaxf(mst[mi][r], mx);
        const float corr = __expf(mst[mi][r] - mnew);
        mst[mi][r] = mnew;
        float rs = 0.0f;
#pragma unroll
        for (int ni = 0; ni < 4; ++ni) {
          const float p = __expf(s[ni][r] - mnew);
          s[ni][r] = p;
          rs += p;
        }
#pragma unroll
        for (int off = 1; off < 16; off <<= 1) rs += __shfl_xor(rs, off, 32);
        lst[mi][r] = lst[mi][r] * corr + rs;
#pragma unroll
        for (int nj = 0; nj < 4; ++nj) o_acc[mi][nj][r] *= corr;
      }
      // P tile -> LDS (C layout -> row-major) for reload in A layout.
      // Same-wave LDS ops are in-order (DScnt), no barrier needed.
#pragma unroll
      for (int ni = 0; ni < 4; ++ni)
#pragma unroll
        for (int r = 0; r < 8; ++r)
          P[(mi * 16 + rbase + r) * 64 + ni * 16 + coln] = (bf16)s[ni][r];
    }

    // O += P @ V
#pragma unroll
    for (int ki = 0; ki < 2; ++ki)
      for (int mi = 0; mi < 4; ++mi) {
        v16bf a = load_frag(P + (mi * 16) * 64 + ki * 32, 64);
#pragma unroll
        for (int nj = 0; nj < 4; ++nj)
          o_acc[mi][nj] = wmma_bf16(a, vf[ki][nj], o_acc[mi][nj]);
      }
  }

  // normalize and write [b, t, h*hd]
#pragma unroll
  for (int mi = 0; mi < 4; ++mi)
#pragma unroll
    for (int r = 0; r < 8; ++r) {
      const float inv = 1.0f / lst[mi][r];
      const int t = q0 + mi * 16 + rbase + r;
#pragma unroll
      for (int nj = 0; nj < 4; ++nj)
        O[(size_t)(bb * SEQ + t) * DIMSZ + h * HD + nj * 16 + coln] =
            (bf16)(o_acc[mi][nj][r] * inv);
    }
}

// ---- Output projection: [BT,D] x [D,D]^T + bias -> f32 ---------------------
__global__ __launch_bounds__(256) void mhsa_out_gemm(
    const bf16* __restrict__ A, const bf16* __restrict__ W,
    const float* __restrict__ bias, float* __restrict__ out) {
  const int NT   = DIMSZ / 64;  // 16
  const int wave = (blockIdx.x * blockDim.x + threadIdx.x) >> 5;
  if (wave >= (BT / 64) * NT) return;
  const int m0 = (wave / NT) * 64, n0 = (wave % NT) * 64;
  const int lane = threadIdx.x & 31;
  const int coln = lane & 15, rbase = (lane >> 4) * 8;

  v8f acc[4][4] = {};
  for (int k = 0; k < DIMSZ; k += 32) {
    v16bf af[4], bfg[4];
#pragma unroll
    for (int i = 0; i < 4; ++i)
      af[i] = load_frag(A + (size_t)(m0 + i * 16) * DIMSZ + k, DIMSZ);
#pragma unroll
    for (int j = 0; j < 4; ++j)
      bfg[j] = load_frag(W + (size_t)(n0 + j * 16) * DIMSZ + k, DIMSZ);
#pragma unroll
    for (int i = 0; i < 4; ++i)
#pragma unroll
      for (int j = 0; j < 4; ++j)
        acc[i][j] = wmma_bf16(af[i], bfg[j], acc[i][j]);
  }
#pragma unroll
  for (int i = 0; i < 4; ++i)
#pragma unroll
    for (int j = 0; j < 4; ++j) {
      const int n = n0 + j * 16 + coln;
      const float bv = bias[n];
#pragma unroll
      for (int r = 0; r < 8; ++r) {
        const int m = m0 + i * 16 + rbase + r;
        out[(size_t)m * DIMSZ + n] = acc[i][j][r] + bv;
      }
    }
}

// ---------------------------------------------------------------------------
extern "C" void kernel_launch(void* const* d_in, const int* in_sizes, int n_in,
                              void* d_out, int out_size, void* d_ws,
                              size_t ws_size, hipStream_t stream) {
  const float* x     = (const float*)d_in[0];
  const float* mask  = (const float*)d_in[1];
  const float* w_qkv = (const float*)d_in[2];
  const float* b_qkv = (const float*)d_in[3];
  const float* w_out = (const float*)d_in[4];
  const float* b_out = (const float*)d_in[5];
  float* out = (float*)d_out;

  char* ws = (char*)d_ws;
  size_t off = 0;
  auto carve = [&](size_t elems) {
    void* p = ws + off;
    off += (elems * sizeof(bf16) + 255) & ~(size_t)255;
    return (bf16*)p;
  };
  bf16* xb  = carve((size_t)BT * DIMSZ);
  bf16* wqb = carve((size_t)ND * DIMSZ);
  bf16* wob = carve((size_t)DIMSZ * DIMSZ);
  bf16* Qb  = carve((size_t)BATCH * NHEAD * SEQ * HD);
  bf16* Kb  = carve((size_t)BATCH * NHEAD * SEQ * HD);
  bf16* Vtb = carve((size_t)BATCH * NHEAD * HD * SEQ);
  bf16* Ob  = carve((size_t)BT * DIMSZ);

  mhsa_cast_bf16<<<2048, 256, 0, stream>>>(x, xb, BT * DIMSZ);
  mhsa_cast_bf16<<<2048, 256, 0, stream>>>(w_qkv, wqb, ND * DIMSZ);
  mhsa_cast_bf16<<<1024, 256, 0, stream>>>(w_out, wob, DIMSZ * DIMSZ);

  // 64x48 wave tiles, 8 waves / block
  mhsa_qkv_gemm<<<(BT / 64) * (ND / 64) / 8, 256, 0, stream>>>(
      xb, wqb, b_qkv, Qb, Kb, Vtb);

  // B*H*(SEQ/64) = 1024 waves
  mhsa_flash_attn<<<BATCH * NHEAD * (SEQ / 64) / 8, 256, 0, stream>>>(
      Qb, Kb, Vtb, mask, Ob);

  // 64x16 wave tiles
  mhsa_out_gemm<<<(BT / 64) * (DIMSZ / 64) / 8, 256, 0, stream>>>(
      Ob, wob, b_out, out);
}
